// MultiHeadedAttention_25177098289531
// MI455X (gfx1250) — matrix operand order repaired and checked by reading, not compiled, not run on hardware
//
#include <hip/hip_runtime.h>
#include <hip/hip_bf16.h>
#include <stdint.h>

typedef __attribute__((ext_vector_type(16))) __bf16          v16bf;
typedef __attribute__((ext_vector_type(8)))  float           v8f;
typedef __attribute__((ext_vector_type(8)))  unsigned short  u16x8;
typedef __attribute__((ext_vector_type(16))) unsigned short  u16x16;
typedef __attribute__((ext_vector_type(4)))  unsigned int    ui32x4;
typedef __attribute__((ext_vector_type(8)))  int             i32x8;
typedef __attribute__((ext_vector_type(4)))  int             i32x4;

#define DM 4096   // d_model == new_d_model
#define NCOL 64
#define NH 8
#define DK 8

// ---------------- helpers ----------------
__device__ __forceinline__ unsigned short f2bf(float f) {
    unsigned int u = __float_as_uint(f);
    unsigned int r = u + 0x7FFFu + ((u >> 16) & 1u);   // round-to-nearest-even
    return (unsigned short)(r >> 16);
}
__device__ __forceinline__ float bf2f(unsigned short h) {
    return __uint_as_float(((unsigned int)h) << 16);
}

// ---------------- fp32 -> bf16 convert ----------------
__global__ __launch_bounds__(256) void f32_to_bf16_k(const float* __restrict__ src,
                                                     unsigned short* __restrict__ dst,
                                                     int n) {
    int i = (blockIdx.x * 256 + threadIdx.x) * 4;
    if (i + 3 < n) {
        float4 f = *(const float4*)(src + i);
        dst[i + 0] = f2bf(f.x);
        dst[i + 1] = f2bf(f.y);
        dst[i + 2] = f2bf(f.z);
        dst[i + 3] = f2bf(f.w);
    }
}

// ---------------- TDM: DMA one 128x32 bf16 tile (row stride 4096) into LDS ----------------
#if __has_builtin(__builtin_amdgcn_tensor_load_to_lds)
#define HAVE_TDM 1
__device__ __forceinline__ void tdm_load_tile(unsigned lds_off, const unsigned short* gptr) {
    unsigned long long ga = (unsigned long long)(uintptr_t)gptr;
    ui32x4 g0;
    g0[0] = 1u;                                   // count=1 (valid user descriptor)
    g0[1] = lds_off;                              // lds_addr (bytes)
    g0[2] = (unsigned)ga;                         // global_addr[31:0]
    g0[3] = (unsigned)(ga >> 32) | 0x80000000u;   // global_addr[56:32] | type=2 (bits 127:126)
    i32x8 g1;
    g1[0] = 0x00010000;                           // workgroup_mask=0, data_size=1 (2 bytes)
    g1[1] = (int)(4096u << 16);                   // tensor_dim0 = 4096 (bits 79:48, low half)
    g1[2] = (int)(4096u << 16);                   // dim0 hi=0 | tensor_dim1 = 4096 (low half)
    g1[3] = (int)(32u << 16);                     // dim1 hi=0 | tile_dim0 = 32
    g1[4] = 128;                                  // tile_dim1 = 128, tile_dim2 = 0
    g1[5] = 4096;                                 // tensor_dim0_stride[31:0] = 4096
    g1[6] = 0;                                    // stride hi / tensor_dim1_stride lo
    g1[7] = 0;
    i32x4 gz4 = {0, 0, 0, 0};                     // groups 2/3 unused (2D tensor)
    i32x8 gz8 = {0, 0, 0, 0, 0, 0, 0, 0};         // extra group (6-arg toolchain form)
    __builtin_amdgcn_tensor_load_to_lds(g0, g1, gz4, gz4, gz8, 0);
}
#else
#define HAVE_TDM 0
#endif

// ---------------- bf16 WMMA GEMM: C[M,N] = A[M,K] * W[N,K]^T + bias ----------------
// WG = 256 thr = 8 waves; 128x128 tile; K-step 32; wave = 64(M) x 32(N) via 4x2 WMMA tiles.
// Tiles staged to LDS by the Tensor Data Mover (wave 0), double-buffered on TENSORcnt.
template<bool OUT_BF16>
__global__ __launch_bounds__(256) void gemm4k_bf16(const unsigned short* __restrict__ A,
                                                   const unsigned short* __restrict__ W,
                                                   const float* __restrict__ bias,
                                                   unsigned short* __restrict__ outB,
                                                   float* __restrict__ outF) {
    __shared__ alignas(128) unsigned short As[2][128 * 32];
    __shared__ alignas(128) unsigned short Bs[2][128 * 32];

    const int tid  = threadIdx.x;
    const int lane = tid & 31;
    const int w    = tid >> 5;
    const int wm   = w >> 2;        // 0..1 -> M offset wm*64
    const int wn   = w & 3;         // 0..3 -> N offset wn*32
    const int mtile = blockIdx.y * 128;
    const int ntile = blockIdx.x * 128;

    v8f acc[4][2];
#pragma unroll
    for (int i = 0; i < 4; ++i)
#pragma unroll
        for (int j = 0; j < 2; ++j) acc[i][j] = (v8f){};

    const int nIter = DM / 32;

#if HAVE_TDM
    if (w == 0) {   // wave 0 owns the DMA pipeline
        tdm_load_tile((unsigned)(uintptr_t)&As[0][0], &A[(size_t)mtile * DM]);
        tdm_load_tile((unsigned)(uintptr_t)&Bs[0][0], &W[(size_t)ntile * DM]);
    }
#else
    const int r0 = tid >> 2,          c0 = (tid & 3) * 8;
    const int r1 = (tid + 256) >> 2,  c1 = ((tid + 256) & 3) * 8;
#endif

    for (int it = 0; it < nIter; ++it) {
        const int cur = it & 1;
        const int kb  = it * 32;
#if HAVE_TDM
        if (w == 0) __builtin_amdgcn_s_wait_tensorcnt(0);
        __syncthreads();                      // publish buffer `cur`; all reads of cur^1 done
        if (w == 0 && it + 1 < nIter) {       // prefetch next K-slice into the other buffer
            const int kn = kb + 32;
            tdm_load_tile((unsigned)(uintptr_t)&As[cur ^ 1][0], &A[(size_t)mtile * DM + kn]);
            tdm_load_tile((unsigned)(uintptr_t)&Bs[cur ^ 1][0], &W[(size_t)ntile * DM + kn]);
        }
#else
        __syncthreads();
        *(u16x8*)&As[cur][r0 * 32 + c0] = *(const u16x8*)&A[(size_t)(mtile + r0) * DM + kb + c0];
        *(u16x8*)&As[cur][r1 * 32 + c1] = *(const u16x8*)&A[(size_t)(mtile + r1) * DM + kb + c1];
        *(u16x8*)&Bs[cur][r0 * 32 + c0] = *(const u16x8*)&W[(size_t)(ntile + r0) * DM + kb + c0];
        *(u16x8*)&Bs[cur][r1 * 32 + c1] = *(const u16x8*)&W[(size_t)(ntile + r1) * DM + kb + c1];
        __syncthreads();
#endif

        // B fragments (ISA layout: lane = N%16; VGPR0-3 hold K {0..7|8..15}, VGPR4-7 hold +16)
        v16bf bfrag[2];
#pragma unroll
        for (int j = 0; j < 2; ++j) {
            const int ncol = wn * 32 + j * 16 + (lane & 15);
            const int koff = (lane >> 4) * 8;
            u16x8 b0 = *(const u16x8*)&Bs[cur][ncol * 32 + koff];
            u16x8 b1 = *(const u16x8*)&Bs[cur][ncol * 32 + koff + 16];
            u16x16 bv = __builtin_shufflevector(b0, b1, 0, 1, 2, 3, 4, 5, 6, 7,
                                                        8, 9, 10, 11, 12, 13, 14, 15);
            bfrag[j] = __builtin_bit_cast(v16bf, bv);
        }
#pragma unroll
        for (int i = 0; i < 4; ++i) {
            // A fragment (ISA layout: lane = M%16; VGPR0-3 hold K {0..7|8..15}, VGPR4-7 hold +16)
            const int mrow = wm * 64 + i * 16 + (lane & 15);
            const int koff = (lane >> 4) * 8;
            u16x8 a0 = *(const u16x8*)&As[cur][mrow * 32 + koff];
            u16x8 a1 = *(const u16x8*)&As[cur][mrow * 32 + koff + 16];
            u16x16 av = __builtin_shufflevector(a0, a1, 0, 1, 2, 3, 4, 5, 6, 7,
                                                        8, 9, 10, 11, 12, 13, 14, 15);
            v16bf afrag = __builtin_bit_cast(v16bf, av);
#pragma unroll
            for (int j = 0; j < 2; ++j) {
                acc[i][j] = __builtin_amdgcn_wmma_f32_16x16x32_bf16(
                    false, afrag, false, bfrag[j], (short)0, acc[i][j], false, false);
            }
        }
    }

    // epilogue: C/D layout -> lane N = l%16; VGPR v -> M = v (+8 for upper half-wave)
#pragma unroll
    for (int i = 0; i < 4; ++i) {
#pragma unroll
        for (int j = 0; j < 2; ++j) {
            const int nOut  = ntile + wn * 32 + j * 16 + (lane & 15);
            const float bvl = bias[nOut];
            const int mBase = mtile + wm * 64 + i * 16 + ((lane >> 4) ? 8 : 0);
#pragma unroll
            for (int v = 0; v < 8; ++v) {
                const float r = acc[i][j][v] + bvl;
                const size_t off = (size_t)(mBase + v) * DM + nOut;
                if (OUT_BF16) outB[off] = f2bf(r);
                else          outF[off] = r;
            }
        }
    }
}

// ---------------- attention: one wave per (batch, head), online softmax ----------------
// q[b,h,c,d] lives at Q[b*4096 + c*64 + h*8 + d]; ctx written back in same layout.
__global__ __launch_bounds__(256) void attn_k(const unsigned short* __restrict__ Q,
                                              const unsigned short* __restrict__ K,
                                              const unsigned short* __restrict__ V,
                                              unsigned short* __restrict__ Ctx) {
    const int lane = threadIdx.x & 31;
    const int wave = threadIdx.x >> 5;
    const int pair = blockIdx.x * 8 + wave;     // 4096 blocks * 8 waves = 32768 (b,h)
    const int b = pair >> 3;
    const int h = pair & 7;
    const size_t base = (size_t)b * DM + h * DK;

    const int rA = lane;        // query rows handled by this lane
    const int rB = lane + 32;
    const float scale = 0.35355339059327373f;   // 1/sqrt(8)

    float qA[DK], qB[DK], accA[DK], accB[DK];
#pragma unroll
    for (int d = 0; d < DK; ++d) {
        qA[d] = bf2f(Q[base + (size_t)rA * NCOL + d]);
        qB[d] = bf2f(Q[base + (size_t)rB * NCOL + d]);
        accA[d] = 0.f; accB[d] = 0.f;
    }
    float mA = -INFINITY, lA = 0.f, mB = -INFINITY, lB = 0.f;

    for (int c = 0; c < NCOL; ++c) {
        float kc[DK], vc[DK];
#pragma unroll
        for (int d = 0; d < DK; ++d) {
            kc[d] = bf2f(K[base + (size_t)c * NCOL + d]);
            vc[d] = bf2f(V[base + (size_t)c * NCOL + d]);
        }
        float sA = 0.f, sB = 0.f;
#pragma unroll
        for (int d = 0; d < DK; ++d) { sA += qA[d] * kc[d]; sB += qB[d] * kc[d]; }
        sA *= scale; sB *= scale;

        float nmA = fmaxf(mA, sA);
        float cA  = __expf(mA - nmA);
        float pA  = __expf(sA - nmA);
        lA = lA * cA + pA;
        float nmB = fmaxf(mB, sB);
        float cB  = __expf(mB - nmB);
        float pB  = __expf(sB - nmB);
        lB = lB * cB + pB;
#pragma unroll
        for (int d = 0; d < DK; ++d) {
            accA[d] = accA[d] * cA + pA * vc[d];
            accB[d] = accB[d] * cB + pB * vc[d];
        }
        mA = nmA; mB = nmB;
    }
    const float invA = 1.f / lA, invB = 1.f / lB;
#pragma unroll
    for (int d = 0; d < DK; ++d) {
        Ctx[base + (size_t)rA * NCOL + d] = f2bf(accA[d] * invA);
        Ctx[base + (size_t)rB * NCOL + d] = f2bf(accB[d] * invB);
    }
}

// ---------------- host launcher ----------------
extern "C" void kernel_launch(void* const* d_in, const int* in_sizes, int n_in,
                              void* d_out, int out_size, void* d_ws, size_t ws_size,
                              hipStream_t stream) {
    (void)in_sizes; (void)n_in; (void)out_size; (void)ws_size;
    const float* x  = (const float*)d_in[0];
    const float* Wq = (const float*)d_in[1];
    const float* bq = (const float*)d_in[2];
    const float* Wk = (const float*)d_in[3];
    const float* bk = (const float*)d_in[4];
    const float* Wv = (const float*)d_in[5];
    const float* bv = (const float*)d_in[6];
    const float* Wo = (const float*)d_in[7];
    const float* bo = (const float*)d_in[8];
    float* out = (float*)d_out;

    const size_t NE = (size_t)DM * DM;           // 16,777,216 elements
    unsigned short* xb  = (unsigned short*)d_ws; // 9 bf16 arrays = 288 MB workspace
    unsigned short* Wqb = xb  + NE;
    unsigned short* Wkb = Wqb + NE;
    unsigned short* Wvb = Wkb + NE;
    unsigned short* Wob = Wvb + NE;
    unsigned short* Qb  = Wob + NE;
    unsigned short* Kb  = Qb  + NE;
    unsigned short* Vb  = Kb  + NE;
    unsigned short* Cb  = Vb  + NE;

    const int nconv = (int)(NE / (256 * 4));     // 16384 blocks
    f32_to_bf16_k<<<nconv, 256, 0, stream>>>(x,  xb,  (int)NE);
    f32_to_bf16_k<<<nconv, 256, 0, stream>>>(Wq, Wqb, (int)NE);
    f32_to_bf16_k<<<nconv, 256, 0, stream>>>(Wk, Wkb, (int)NE);
    f32_to_bf16_k<<<nconv, 256, 0, stream>>>(Wv, Wvb, (int)NE);
    f32_to_bf16_k<<<nconv, 256, 0, stream>>>(Wo, Wob, (int)NE);

    dim3 ggrid(DM / 128, DM / 128), gblk(256);
    gemm4k_bf16<true ><<<ggrid, gblk, 0, stream>>>(xb, Wqb, bq, Qb, nullptr);
    gemm4k_bf16<true ><<<ggrid, gblk, 0, stream>>>(xb, Wkb, bk, Kb, nullptr);
    gemm4k_bf16<true ><<<ggrid, gblk, 0, stream>>>(xb, Wvb, bv, Vb, nullptr);

    attn_k<<<DM, 256, 0, stream>>>(Qb, Kb, Vb, Cb);

    gemm4k_bf16<false><<<ggrid, gblk, 0, stream>>>(Cb, Wob, bo, nullptr, out);
}
